// LieTransformerBlock_37881611551186
// MI455X (gfx1250) — compile-verified, matrix operand order checked by reading, not run on hardware
//
#include <hip/hip_runtime.h>
#include <hip/hip_bf16.h>
#include <math.h>
#include <stdint.h>

typedef __attribute__((ext_vector_type(16))) _Float16 v16h;
typedef __attribute__((ext_vector_type(8)))  float    v8f;

// ---- problem constants (from reference setup) ----
#define BB    4
#define TT    4
#define HH    16
#define WW    16
#define CC    128
#define HEADS 8
#define DD    16
#define NN    256          // H*W
#define NTOT  1024         // T*N
#define ROWS  (BB*NTOT)    // 4096 tokens
#define HID   340
#define HID2  680

// WMMA fragment K index for 16-bit A/B fragments (ISA 7.12.2):
//   lanes 0-15 : VGPR r -> K = 2r,2r+1 (r<4) else 16+2(r-4),...
//   lanes16-31 : same rows, K shifted by +8 within each 16-block
__device__ __forceinline__ int frag_k(int k0, int e, int halfid) {
    return k0 + e + 8 * halfid + 8 * (e >> 3);
}

__device__ __forceinline__ v8f wmma_f16(const v16h& a, const v16h& b, const v8f& c) {
    return __builtin_amdgcn_wmma_f32_16x16x32_f16(false, a, false, b, (short)0, c, false, false);
}

// =====================================================================
// K0/K6: LayerNorm over C=128 (one block of 128 threads per token row)
// =====================================================================
__global__ void __launch_bounds__(128)
ln_kernel(const float* __restrict__ x, const float* __restrict__ s,
          const float* __restrict__ b, _Float16* __restrict__ out) {
    int row = blockIdx.x, tid = threadIdx.x;
    float v = x[(long)row * CC + tid];
    __shared__ float red[128];
    red[tid] = v; __syncthreads();
    for (int o = 64; o > 0; o >>= 1) { if (tid < o) red[tid] += red[tid + o]; __syncthreads(); }
    float mean = red[0] * (1.0f / CC); __syncthreads();
    float d = v - mean;
    red[tid] = d * d; __syncthreads();
    for (int o = 64; o > 0; o >>= 1) { if (tid < o) red[tid] += red[tid + o]; __syncthreads(); }
    float var = red[0] * (1.0f / CC);
    float xn = d * rsqrtf(var + 1e-5f) * s[tid] + b[tid];
    out[(long)row * CC + tid] = (_Float16)xn;
}

// =====================================================================
// K1: QKV projection  (4096x128) @ (128x384)  -> scatter to q/k/v [bh,tok,d]
// =====================================================================
__global__ void __launch_bounds__(128)
qkv_kernel(const _Float16* __restrict__ xn, const float* __restrict__ w,
           const float* __restrict__ wb,
           _Float16* __restrict__ q, _Float16* __restrict__ k, _Float16* __restrict__ v) {
    int wave = threadIdx.x >> 5;
    int tm = blockIdx.x * 4 + wave;           // 0..255
    int tn = blockIdx.y;                      // 0..23
    int lane = threadIdx.x & 31, halfid = lane >> 4, r = lane & 15;
    const _Float16* A = xn + (long)(tm * 16) * CC;
    v8f acc = {};
    for (int k0 = 0; k0 < CC; k0 += 32) {
        v16h a, bf;
#pragma unroll
        for (int e = 0; e < 16; ++e) {
            int kk = frag_k(k0, e, halfid);
            a[e]  = A[(long)r * CC + kk];
            bf[e] = (_Float16)w[(long)(tn * 16 + r) * CC + kk];   // B[k][n] = qkv_w[n][k]
        }
        acc = wmma_f16(a, bf, acc);
    }
#pragma unroll
    for (int vr = 0; vr < 8; ++vr) {
        int m = tm * 16 + vr + 8 * halfid;    // token row (b*1024+tok)
        int n = tn * 16 + r;                  // 0..383
        float val = acc[vr] + wb[n];
        int which = n >> 7, hh = (n & 127) >> 4, dd = n & 15;
        int bb = m >> 10, tok = m & 1023;
        long idx = (((long)(bb * HEADS + hh)) * NTOT + tok) * DD + dd;
        _Float16 hv = (_Float16)val;
        if (which == 0) q[idx] = hv; else if (which == 1) k[idx] = hv; else v[idx] = hv;
    }
}

// =====================================================================
// K2: logits = (q @ k^T) * d^-0.5 + biasMLP(bias)   (fused, written to d_out)
// =====================================================================
__global__ void __launch_bounds__(128)
qk_bias_kernel(const _Float16* __restrict__ q, const _Float16* __restrict__ kmat,
               const float* __restrict__ bias,
               const float* __restrict__ bw1, const float* __restrict__ bb1,
               const float* __restrict__ bw2, const float* __restrict__ bb2,
               float* __restrict__ attn) {
    int wave = threadIdx.x >> 5;
    int bh = blockIdx.z; int bb = bh >> 3; int h = bh & 7;
    int tm = blockIdx.x;                      // 0..63
    int tn = blockIdx.y * 4 + wave;           // 0..63
    int lane = threadIdx.x & 31, halfid = lane >> 4, r = lane & 15;
    const _Float16* A  = q    + ((long)bh * NTOT + tm * 16) * DD;
    const _Float16* Bq = kmat + ((long)bh * NTOT + tn * 16) * DD;
    v16h a, bf;
#pragma unroll
    for (int e = 0; e < 16; ++e) {
        int kd = frag_k(0, e, halfid);        // K dim = d = 16, pad 16..31 with zero
        a[e]  = (kd < DD) ? A[(long)r * DD + kd]  : (_Float16)0.0f;
        bf[e] = (kd < DD) ? Bq[(long)r * DD + kd] : (_Float16)0.0f; // B[d][n] = k[n][d]
    }
    v8f acc = {};
    acc = wmma_f16(a, bf, acc);

    float w1[16], b1[16], w2[16];
#pragma unroll
    for (int t = 0; t < 16; ++t) { w1[t] = bw1[t]; b1[t] = bb1[t]; w2[t] = bw2[h * 16 + t]; }
    float b2 = bb2[h];
#pragma unroll
    for (int vr = 0; vr < 8; ++vr) {
        int m = tm * 16 + vr + 8 * halfid;
        int n = tn * 16 + r;
        float bv = bias[((long)bb * NTOT + m) * NTOT + n];
        float lb = b2;
#pragma unroll
        for (int t = 0; t < 16; ++t) {
            float u = fmaf(bv, w1[t], b1[t]);
            lb = fmaf(u > 0.0f ? u : 0.0f, w2[t], lb);
        }
        attn[(((long)bh) * NTOT + m) * NTOT + n] = acc[vr] * 0.25f + lb;  // 16^-0.5
    }
}

// =====================================================================
// K3+K4 fused: softmax (in place, fp32 to d_out) + attn@v via WMMA.
//   Block = 256 threads = 16 rows x 16 threads; owns one 16-row tile.
//   - v (32KB per (b,h)) staged to LDS with global_load_async_to_lds_b128
//     issued at kernel entry; copy overlaps the whole softmax phase and
//     is fenced with s_wait_asynccnt right before the WMMA sweep.
//   - logits held in registers (64 fp32/thread) for max/exp/sum
//   - normalized fp32 attn written to d_out
//   - UNnormalized exp() stored as f16 in LDS; WMMA K-sweep split over
//     8 waves; partial 16x16 tiles reduced through LDS; 1/sum applied
//     to the final f32 accumulators (softmax normalization is linear).
// =====================================================================
__global__ void __launch_bounds__(256)
softmax_av_kernel(float* __restrict__ attn, const _Float16* __restrict__ v,
                  _Float16* __restrict__ aout) {
    __shared__ __align__(16) _Float16 sA[16 * NTOT];   // 32 KB: exp(logit-max), f16
    __shared__ __align__(16) _Float16 sV[NTOT * DD];   // 32 KB: v[tok][d] staged
    __shared__ float    sPart[8 * 256];                // 8 KB: per-wave partial tiles
    __shared__ float    sRed[256];
    __shared__ float    sInv[16];

    int bh = blockIdx.z; int bb = bh >> 3; int h = bh & 7;
    int tm = blockIdx.x;                  // 0..63
    int tid = threadIdx.x;
    int rl = tid >> 4, sub = tid & 15;    // row-local 0..15, col-group 0..15

    // ---- issue async bulk copy of v -> LDS (overlaps softmax below) ----
    {
        uint32_t ldsbase = (uint32_t)(uintptr_t)&sV[0];
        const char* gbase = (const char*)(v + (long)bh * NTOT * DD);
#pragma unroll
        for (int it = 0; it < 8; ++it) {
            uint32_t ldsa = ldsbase + (uint32_t)(tid * 16 + it * 4096);
            const char* ga = gbase + tid * 16 + it * 4096;
            asm volatile("global_load_async_to_lds_b128 %0, %1, off"
                         :: "v"(ldsa), "v"(ga) : "memory");
        }
    }

    float* rowp = attn + ((long)bh * NTOT + tm * 16 + rl) * NTOT;

    // ---- pass 1: load row chunk into registers, local max ----
    float ev[64];
    float mx = -1e30f;
#pragma unroll
    for (int j = 0; j < 64; ++j) { ev[j] = rowp[sub + 16 * j]; mx = fmaxf(mx, ev[j]); }
    sRed[tid] = mx; __syncthreads();
    if (sub < 8) sRed[tid] = fmaxf(sRed[tid], sRed[tid + 8]); __syncthreads();
    if (sub < 4) sRed[tid] = fmaxf(sRed[tid], sRed[tid + 4]); __syncthreads();
    if (sub < 2) sRed[tid] = fmaxf(sRed[tid], sRed[tid + 2]); __syncthreads();
    if (sub < 1) sRed[tid] = fmaxf(sRed[tid], sRed[tid + 1]); __syncthreads();
    mx = sRed[rl * 16]; __syncthreads();

    // ---- pass 2: exp, stash f16 in LDS, local sum ----
    float s = 0.0f;
#pragma unroll
    for (int j = 0; j < 64; ++j) {
        int cc = sub + 16 * j;
        float e = expf(ev[j] - mx);
        ev[j] = e; s += e;
        sA[rl * NTOT + cc] = (_Float16)e;
    }
    sRed[tid] = s; __syncthreads();
    if (sub < 8) sRed[tid] += sRed[tid + 8]; __syncthreads();
    if (sub < 4) sRed[tid] += sRed[tid + 4]; __syncthreads();
    if (sub < 2) sRed[tid] += sRed[tid + 2]; __syncthreads();
    if (sub < 1) sInv[rl] = 1.0f / (sRed[tid] + sRed[tid + 1]);
    __syncthreads();
    float inv = sInv[rl];

    // ---- pass 3: normalized fp32 attn back to d_out ----
#pragma unroll
    for (int j = 0; j < 64; ++j) rowp[sub + 16 * j] = ev[j] * inv;

    // ---- fence the async v copy, make LDS visible to all waves ----
    asm volatile("s_wait_asynccnt 0x0" ::: "memory");
    __syncthreads();

    // ---- WMMA phase: wave w covers K in [w*128, w*128+128) ----
    int wave = tid >> 5, lane = tid & 31, halfid = lane >> 4, r = lane & 15;
    v8f acc = {};
    for (int k0 = wave * 128; k0 < wave * 128 + 128; k0 += 32) {
        v16h a, bf;
#pragma unroll
        for (int e = 0; e < 16; ++e) {
            int kk = frag_k(k0, e, halfid);
            a[e]  = sA[r * NTOT + kk];        // ds loads from LDS
            bf[e] = sV[kk * DD + r];          // B[k=tok][n=d] from LDS
        }
        acc = wmma_f16(a, bf, acc);
    }
#pragma unroll
    for (int vr = 0; vr < 8; ++vr) sPart[wave * 256 + vr * 32 + lane] = acc[vr];
    __syncthreads();

    // ---- cross-wave reduce + row-scale + store (one element / thread) ----
    float tot = 0.0f;
#pragma unroll
    for (int w2 = 0; w2 < 8; ++w2) tot += sPart[w2 * 256 + tid];
    int vr2 = tid >> 5, l2 = tid & 31;
    int M = vr2 + 8 * (l2 >> 4), Nn = l2 & 15;
    float o = tot * sInv[M];
    int m = tm * 16 + M;
    aout[((long)bb * NTOT + m) * CC + h * DD + Nn] = (_Float16)o;
}

// =====================================================================
// K5: proj + residual:  x1 = x + gamma1 * (aout @ proj_w^T + proj_b)
// =====================================================================
__global__ void __launch_bounds__(128)
proj_kernel(const _Float16* __restrict__ aout, const float* __restrict__ w,
            const float* __restrict__ wb, const float* __restrict__ x,
            const float* __restrict__ gamma1, float* __restrict__ x1) {
    int wave = threadIdx.x >> 5;
    int tm = blockIdx.x * 4 + wave;           // 0..255
    int tn = blockIdx.y;                      // 0..7
    int lane = threadIdx.x & 31, halfid = lane >> 4, r = lane & 15;
    const _Float16* A = aout + (long)(tm * 16) * CC;
    v8f acc = {};
    for (int k0 = 0; k0 < CC; k0 += 32) {
        v16h a, bf;
#pragma unroll
        for (int e = 0; e < 16; ++e) {
            int kk = frag_k(k0, e, halfid);
            a[e]  = A[(long)r * CC + kk];
            bf[e] = (_Float16)w[(long)(tn * 16 + r) * CC + kk];
        }
        acc = wmma_f16(a, bf, acc);
    }
#pragma unroll
    for (int vr = 0; vr < 8; ++vr) {
        int m = tm * 16 + vr + 8 * halfid;
        int n = tn * 16 + r;
        long idx = (long)m * CC + n;
        x1[idx] = x[idx] + gamma1[n] * (acc[vr] + wb[n]);
    }
}

// =====================================================================
// K7: 1x1 conv "pin":  pin_out[bt,c,p] from xn2 tokens GEMM, oc=680
// =====================================================================
__global__ void __launch_bounds__(128)
pin_kernel(const _Float16* __restrict__ xn2, const float* __restrict__ w,
           const float* __restrict__ wb, float* __restrict__ pout) {
    int wave = threadIdx.x >> 5;
    int tm = blockIdx.x * 4 + wave;           // 0..255 (token tiles)
    int tn = blockIdx.y;                      // 0..42 (680 -> 43 tiles)
    int lane = threadIdx.x & 31, halfid = lane >> 4, r = lane & 15;
    int ncol = tn * 16 + r;
    const _Float16* A = xn2 + (long)(tm * 16) * CC;
    v8f acc = {};
    for (int k0 = 0; k0 < CC; k0 += 32) {
        v16h a, bf;
#pragma unroll
        for (int e = 0; e < 16; ++e) {
            int kk = frag_k(k0, e, halfid);
            a[e]  = A[(long)r * CC + kk];
            bf[e] = (ncol < HID2) ? (_Float16)w[(long)ncol * CC + kk] : (_Float16)0.0f;
        }
        acc = wmma_f16(a, bf, acc);
    }
#pragma unroll
    for (int vr = 0; vr < 8; ++vr) {
        int m = tm * 16 + vr + 8 * halfid;    // g = bt*256 + p
        int n = tn * 16 + r;
        if (n < HID2) {
            int bt = m >> 8, p = m & 255;
            pout[((long)bt * HID2 + n) * NN + p] = acc[vr] + wb[n];
        }
    }
}

// =====================================================================
// K8: depthwise 3x3 (pad 1) + exact GELU gating -> glu[g][c] f16
// =====================================================================
__global__ void __launch_bounds__(256)
dw_glu_kernel(const float* __restrict__ pin_out, const float* __restrict__ dw_w,
              const float* __restrict__ dw_b, _Float16* __restrict__ glu) {
    long idx = (long)blockIdx.x * blockDim.x + threadIdx.x;
    const long total = (long)BB * TT * HID * NN;   // 16*340*256
    if (idx >= total) return;
    int p = idx & 255;
    int c = (int)((idx >> 8) % HID);
    int bt = (int)(idx / ((long)HID * NN));
    int hh = p >> 4, ww = p & 15;
    float u = dw_b[c], g = dw_b[c + HID];
#pragma unroll
    for (int dy = -1; dy <= 1; ++dy)
#pragma unroll
        for (int dx = -1; dx <= 1; ++dx) {
            int y = hh + dy, x = ww + dx;
            if (y < 0 || y > 15 || x < 0 || x > 15) continue;
            int kidx = (dy + 1) * 3 + (dx + 1);
            float pu = pin_out[((long)bt * HID2 + c) * NN + y * 16 + x];
            float pg = pin_out[((long)bt * HID2 + c + HID) * NN + y * 16 + x];
            u = fmaf(dw_w[(long)c * 9 + kidx], pu, u);
            g = fmaf(dw_w[(long)(c + HID) * 9 + kidx], pg, g);
        }
    float ge = 0.5f * u * (1.0f + erff(u * 0.70710678118f));   // exact GELU
    glu[((long)bt * NN + p) * HID + c] = (_Float16)(ge * g);
}

// =====================================================================
// K9: 1x1 conv "pout" + residual -> final x into d_out.  K=340 (padded)
// =====================================================================
__global__ void __launch_bounds__(128)
pout_kernel(const _Float16* __restrict__ glu, const float* __restrict__ w,
            const float* __restrict__ wb, const float* __restrict__ x1,
            const float* __restrict__ gamma2, float* __restrict__ xout) {
    int wave = threadIdx.x >> 5;
    int tm = blockIdx.x * 4 + wave;           // 0..255
    int tn = blockIdx.y;                      // 0..7
    int lane = threadIdx.x & 31, halfid = lane >> 4, r = lane & 15;
    const _Float16* A = glu + (long)(tm * 16) * HID;
    v8f acc = {};
    for (int k0 = 0; k0 < 352; k0 += 32) {    // 11 steps, last partially padded
        v16h a, bf;
#pragma unroll
        for (int e = 0; e < 16; ++e) {
            int kk = frag_k(k0, e, halfid);
            bool ok = (kk < HID);
            a[e]  = ok ? A[(long)r * HID + kk] : (_Float16)0.0f;
            bf[e] = ok ? (_Float16)w[(long)(tn * 16 + r) * HID + kk] : (_Float16)0.0f;
        }
        acc = wmma_f16(a, bf, acc);
    }
#pragma unroll
    for (int vr = 0; vr < 8; ++vr) {
        int m = tm * 16 + vr + 8 * halfid;
        int n = tn * 16 + r;
        long idx = (long)m * CC + n;
        xout[idx] = x1[idx] + gamma2[n] * (acc[vr] + wb[n]);
    }
}

// =====================================================================
extern "C" void kernel_launch(void* const* d_in, const int* in_sizes, int n_in,
                              void* d_out, int out_size, void* d_ws, size_t ws_size,
                              hipStream_t stream) {
    const float* x      = (const float*)d_in[0];
    const float* bias   = (const float*)d_in[1];
    const float* ln1_s  = (const float*)d_in[2];
    const float* ln1_b  = (const float*)d_in[3];
    const float* qkv_w  = (const float*)d_in[4];
    const float* qkv_b  = (const float*)d_in[5];
    const float* proj_w = (const float*)d_in[6];
    const float* proj_b = (const float*)d_in[7];
    const float* bw1    = (const float*)d_in[8];
    const float* bb1    = (const float*)d_in[9];
    const float* bw2    = (const float*)d_in[10];
    const float* bb2    = (const float*)d_in[11];
    const float* ln2_s  = (const float*)d_in[12];
    const float* ln2_b  = (const float*)d_in[13];
    const float* pin_w  = (const float*)d_in[14];
    const float* pin_b  = (const float*)d_in[15];
    const float* dw_w   = (const float*)d_in[16];
    const float* dw_b   = (const float*)d_in[17];
    const float* pout_w = (const float*)d_in[18];
    const float* pout_b = (const float*)d_in[19];
    const float* gamma1 = (const float*)d_in[20];
    const float* gamma2 = (const float*)d_in[21];

    float* xout_d = (float*)d_out;                               // (B,Ntot,C)
    float* attn_d = (float*)d_out + (size_t)BB * NTOT * CC;      // (B,heads,Ntot,Ntot)

    // ---- workspace layout (bytes, 256B aligned) ----
    char* ws = (char*)d_ws;
    size_t off = 0;
    auto alloc = [&](size_t bytes) { size_t o = off; off = (off + bytes + 255) & ~(size_t)255; return o; };
    _Float16* xn_h   = (_Float16*)(ws + alloc((size_t)ROWS * CC * 2));
    _Float16* q_h    = (_Float16*)(ws + alloc((size_t)BB * HEADS * NTOT * DD * 2));
    _Float16* k_h    = (_Float16*)(ws + alloc((size_t)BB * HEADS * NTOT * DD * 2));
    _Float16* v_h    = (_Float16*)(ws + alloc((size_t)BB * HEADS * NTOT * DD * 2));
    _Float16* aout_h = (_Float16*)(ws + alloc((size_t)ROWS * CC * 2));
    float*    x1     = (float*)   (ws + alloc((size_t)ROWS * CC * 4));
    _Float16* xn2_h  = (_Float16*)(ws + alloc((size_t)ROWS * CC * 2));
    float*    pin_o  = (float*)   (ws + alloc((size_t)ROWS * HID2 * 4));
    _Float16* glu_h  = (_Float16*)(ws + alloc((size_t)ROWS * HID * 2));
    (void)ws_size; (void)in_sizes; (void)n_in; (void)out_size;

    // --- attention branch ---
    ln_kernel<<<ROWS, 128, 0, stream>>>(x, ln1_s, ln1_b, xn_h);
    qkv_kernel<<<dim3(64, 24), 128, 0, stream>>>(xn_h, qkv_w, qkv_b, q_h, k_h, v_h);
    qk_bias_kernel<<<dim3(64, 16, BB * HEADS), 128, 0, stream>>>(
        q_h, k_h, bias, bw1, bb1, bw2, bb2, attn_d);
    softmax_av_kernel<<<dim3(64, 1, BB * HEADS), 256, 0, stream>>>(attn_d, v_h, aout_h);
    proj_kernel<<<dim3(64, 8), 128, 0, stream>>>(aout_h, proj_w, proj_b, x, gamma1, x1);

    // --- GDFN branch ---
    ln_kernel<<<ROWS, 128, 0, stream>>>(x1, ln2_s, ln2_b, xn2_h);
    pin_kernel<<<dim3(64, 43), 128, 0, stream>>>(xn2_h, pin_w, pin_b, pin_o);
    {
        long total = (long)BB * TT * HID * NN;
        int blocks = (int)((total + 255) / 256);
        dw_glu_kernel<<<blocks, 256, 0, stream>>>(pin_o, dw_w, dw_b, glu_h);
    }
    pout_kernel<<<dim3(64, 8), 128, 0, stream>>>(glu_h, pout_w, pout_b, x1, gamma2, xout_d);
}
